// CMMS_GCL_45260365365925
// MI455X (gfx1250) — matrix operand order repaired and checked by reading, compiled, not run on hardware
//
#include <hip/hip_runtime.h>
#include <cstdint>
#include <cstddef>

// ---------------------------------------------------------------------------
// Types for CDNA5 WMMA (wave32, gfx1250)
// ---------------------------------------------------------------------------
typedef __bf16 bf16;
typedef __bf16 v16bf __attribute__((ext_vector_type(16)));
typedef __bf16 v8bf  __attribute__((ext_vector_type(8)));
typedef float  v8f   __attribute__((ext_vector_type(8)));

#define CAT16(lo, hiv) __builtin_shufflevector(lo, hiv, 0,1,2,3,4,5,6,7,8,9,10,11,12,13,14,15)

#define NNODES  100000
#define NEDGES  1600000
#define NGRAPHS 1024
#define FXD     84
#define FXDP    96      // FXD padded to mult of 32
#define FOUT    840
#define HGRU    100
#define SEQ     1024    // scan length (reference scans over the 1024 axis)
#define BGRU    100     // "batch" inside the scan

__device__ __forceinline__ float sigmoidf_(float x) { return 1.0f / (1.0f + __expf(-x)); }

// ---------------------------------------------------------------------------
// Generic WMMA GEMM:  C[M,N] = act( A[M,Kp](bf16,lda) * W[N,Kp](bf16,ldw)^T + bias[N] )
// Kp (=K padded with zeros) must be a multiple of 32; lda/ldw multiples of 8
// and 16B-aligned bases -> every fragment half is one aligned 16B vector load.
// Register-tiled: one wave computes a 16x64 strip (4 accumulators), so the A
// fragment is loaded once per k-step and feeds 4 WMMAs (4x less A traffic).
// ACT: 0=none 1=relu 2=tanh.  OMODE: 0=f32 out, 1=bf16 out, 2=pooled epilogue
// (relu'd values atomically max/sum-pooled per graph via gidx[]).
// Requires M % 16 == 0 (true for every call).
// Block = 256 threads = 8 waves = 16 rows x 512 cols of C.
// Columns n >= N are computed from a clamped W row and simply never stored
// (in WMMA, B column j only feeds D column j) -> branch-free inner loop.
// ---------------------------------------------------------------------------
template<int ACT, int OMODE>
__global__ __launch_bounds__(256) void gemm_bias_act(
    const bf16* __restrict__ A, int lda,
    const bf16* __restrict__ W, int ldw,
    const float* __restrict__ bias, void* __restrict__ Cout, int ldc,
    int M, int N, int Kp,
    const int* __restrict__ gidx, float* __restrict__ pmax,
    float* __restrict__ psum, int pld)
{
  const int lane = threadIdx.x & 31;
  const int wave = threadIdx.x >> 5;
  const int m0 = blockIdx.y * 16;
  const int nbase = (blockIdx.x * 8 + wave) * 64;
  if (nbase >= N) return;              // wave-uniform: EXEC stays all-ones for WMMA
  const int hi  = lane >> 4;           // 0: lanes 0-15, 1: lanes 16-31
  const int l16 = lane & 15;
  const int m = m0 + l16;              // A row this lane feeds (M%16==0 -> valid)

  const bf16* arow = A + (size_t)m * lda;
  const bf16* wrow[4];
#pragma unroll
  for (int j = 0; j < 4; ++j) {
    const int n = nbase + j * 16 + l16;
    const int nc = (n < N) ? n : (N - 1);   // clamped; garbage cols never stored
    wrow[j] = W + (size_t)nc * ldw;
  }

  v8f acc[4];
#pragma unroll
  for (int j = 0; j < 4; ++j) acc[j] = (v8f){0.f,0.f,0.f,0.f,0.f,0.f,0.f,0.f};

  for (int kk = 0; kk < Kp; kk += 32) {
    // A 16-bit 16x32 fragment: lanes0-15 K in {kk..kk+7, kk+16..kk+23};
    // lanes16-31 K in {kk+8..kk+15, kk+24..kk+31}.  (lda%8==0 -> 16B aligned)
    const v8bf* ap = (const v8bf*)(arow + kk + hi * 8);
    const v16bf af = CAT16(ap[0], ap[2]);
    __builtin_prefetch(arow + kk + 64, 0, 3);
#pragma unroll
    for (int j = 0; j < 4; ++j) {
      const v8bf* wp = (const v8bf*)(wrow[j] + kk + hi * 8);
      v16bf bv = CAT16(wp[0], wp[2]);
      acc[j] = __builtin_amdgcn_wmma_f32_16x16x32_bf16(false, af, false, bv,
                                                       (short)0, acc[j], false, false);
    }
  }

  // Epilogue. C layout: lane<16 -> N=cn, VGPR r -> M=m0+r;
  //           lane>=16 -> N=cn, VGPR r -> M=m0+8+r.
#pragma unroll
  for (int j = 0; j < 4; ++j) {
    const int cn = nbase + j * 16 + l16;
    if (cn >= N) continue;
    const float bv0 = bias ? bias[cn] : 0.0f;
#pragma unroll
    for (int r = 0; r < 8; ++r) {
      const int cm = m0 + hi * 8 + r;
      float v = acc[j][r] + bv0;
      if (ACT == 1) v = v > 0.0f ? v : 0.0f;
      if (ACT == 2) v = tanhf(v);
      if (OMODE == 0) {
        ((float*)Cout)[(size_t)cm * ldc + cn] = v;
      } else if (OMODE == 1) {
        ((bf16*)Cout)[(size_t)cm * ldc + cn] = (bf16)v;
      } else {
        const int g = gidx[cm];
        atomicAdd(&psum[(size_t)g * pld + cn], v);
        // v >= 0 after relu -> IEEE order == int order; pool init is 0
        atomicMax((int*)&pmax[(size_t)g * pld + cn], __float_as_int(v));
      }
    }
  }
}

// ---------------------------------------------------------------------------
// Persistent fused bidirectional GRU: grid.x = 2 (one WGP-workgroup per
// direction), 1024 threads, 1024 sequential steps. LDS tiles are padded
// (Whh 304x136 bf16, h 112x136 bf16, gh 112x304 f32) so all WMMA fragment
// loads are unconditional aligned 16B vector loads (ds_load_b128) - no
// per-element predication. gi = X@Wih^T + bih precomputed in global memory.
// ---------------------------------------------------------------------------
#define WHH_STR 136
#define H_STR   136
#define GH_STR  304
#define WHH_OFF 0
#define H_OFF   (304 * WHH_STR * 2)                 // 82,688
#define GH_OFF  (H_OFF + 112 * H_STR * 2)           // 113,152
#define GRU_LDS_BYTES (GH_OFF + 112 * GH_STR * 4)   // 249,344 < 320KB

__global__ __launch_bounds__(1024) void gru_fused(
    const bf16* __restrict__ WhhP_f, const bf16* __restrict__ WhhP_b,  // 300x128 padded
    const float* __restrict__ bhh_f, const float* __restrict__ bhh_b,
    const float* __restrict__ gi_f, const float* __restrict__ gi_b,    // (1024*100,300)
    bf16* __restrict__ ys_f, bf16* __restrict__ ys_b)
{
  extern __shared__ char smem[];
  bf16*  whh_s = (bf16*)(smem + WHH_OFF);
  bf16*  h_s   = (bf16*)(smem + H_OFF);
  float* gh_s  = (float*)(smem + GH_OFF);

  const int rev = blockIdx.x;
  const bf16*  WhhP = rev ? WhhP_b : WhhP_f;
  const float* bhh  = rev ? bhh_b  : bhh_f;
  const float* gi   = rev ? gi_b   : gi_f;
  bf16*        ys   = rev ? ys_b   : ys_f;

  // zero padded tiles, then fill Whh rows (padded source already has zero cols)
  for (int i = threadIdx.x; i < 304 * WHH_STR; i += 1024) whh_s[i] = (bf16)0.0f;
  for (int i = threadIdx.x; i < 112 * H_STR; i += 1024) h_s[i] = (bf16)0.0f;
  __syncthreads();
  for (int i = threadIdx.x; i < 300 * 128; i += 1024)
    whh_s[(i >> 7) * WHH_STR + (i & 127)] = WhhP[i];
  __syncthreads();

  const int lane = threadIdx.x & 31;
  const int wave = threadIdx.x >> 5;
  const int hi   = lane >> 4;
  const int l16  = lane & 15;

  for (int t = 0; t < SEQ; ++t) {
    const int tt = rev ? (SEQ - 1 - t) : t;

    // Phase A: gh = h @ Whh^T   (M=100 -> 7 tiles, N=300 -> 19 tiles, K=128)
    for (int tile = wave; tile < 7 * 19; tile += 32) {
      const int m0 = (tile / 19) * 16;
      const int n0 = (tile % 19) * 16;
      const bf16* hrow = h_s + (m0 + l16) * H_STR;
      const bf16* wr   = whh_s + (n0 + l16) * WHH_STR;
      v8f acc = {0.f, 0.f, 0.f, 0.f, 0.f, 0.f, 0.f, 0.f};
#pragma unroll
      for (int kk = 0; kk < 128; kk += 32) {
        const v8bf* hp = (const v8bf*)(hrow + kk + hi * 8);
        const v8bf* wp = (const v8bf*)(wr + kk + hi * 8);
        v16bf af = CAT16(hp[0], hp[2]);
        v16bf bv = CAT16(wp[0], wp[2]);
        acc = __builtin_amdgcn_wmma_f32_16x16x32_bf16(false, af, false, bv,
                                                      (short)0, acc, false, false);
      }
      const int cn = n0 + l16;        // <= 303: in padded bounds, unguarded
#pragma unroll
      for (int r = 0; r < 8; ++r)
        gh_s[(m0 + hi * 8 + r) * GH_STR + cn] = acc[r];
    }
    __syncthreads();

    // Phase B: gates + state update + write ys[tt]
    for (int id = threadIdx.x; id < BGRU * HGRU; id += 1024) {
      const int b = id / HGRU, j = id % HGRU;
      const float* gir = gi + ((size_t)tt * BGRU + b) * 300;
      const float* ghr = gh_s + b * GH_STR;
      const float r = sigmoidf_(gir[j]       + ghr[j]       + bhh[j]);
      const float z = sigmoidf_(gir[j + 100] + ghr[j + 100] + bhh[j + 100]);
      const float g = tanhf(gir[j + 200] + r * (ghr[j + 200] + bhh[j + 200]));
      const float hp = (float)h_s[b * H_STR + j];
      const float hn = (1.0f - z) * g + z * hp;
      h_s[b * H_STR + j] = (bf16)hn;
      ys[((size_t)tt * BGRU + b) * HGRU + j] = (bf16)hn;
    }
    __syncthreads();
  }
}

// ---------------------------------------------------------------------------
// Elementwise / reduction helpers (all padding-aware)
// ---------------------------------------------------------------------------
__global__ void k_fill_f32(float* p, float v, long long n) {
  long long i = blockIdx.x * (long long)blockDim.x + threadIdx.x;
  if (i < n) p[i] = v;
}
// dst[row*Kp + col] = col < K ? (bf16)src[row*K + col] : 0
__global__ void k_cvt_pad(const float* __restrict__ s, bf16* __restrict__ d,
                          long long rows, int K, int Kp) {
  long long i = blockIdx.x * (long long)blockDim.x + threadIdx.x;
  if (i >= rows * Kp) return;
  long long r = i / Kp; int c = (int)(i % Kp);
  d[i] = (c < K) ? (bf16)s[r * K + c] : (bf16)0.0f;
}
// dst[row*96+col] = col<84 ? (bf16)(a+b) : 0
__global__ void k_add_pad(const float* __restrict__ a, const float* __restrict__ b,
                          bf16* __restrict__ d, long long rows) {
  long long i = blockIdx.x * (long long)blockDim.x + threadIdx.x;
  if (i >= rows * FXDP) return;
  long long r = i / FXDP; int c = (int)(i % FXDP);
  d[i] = (c < FXD) ? (bf16)(a[r * FXD + c] + b[r * FXD + c]) : (bf16)0.0f;
}
// hcat = relu(concat(fwd,bwd)) -> (102400, 224) bf16, cols 200..223 zero
__global__ void k_hcat(const bf16* __restrict__ f, const bf16* __restrict__ b,
                       bf16* __restrict__ o, long long rows) {
  long long i = blockIdx.x * (long long)blockDim.x + threadIdx.x;
  if (i >= rows * 224) return;
  long long r = i / 224; int j = (int)(i % 224);
  float v = 0.f;
  if (j < 100) v = (float)f[r * 100 + j];
  else if (j < 200) v = (float)b[r * 100 + (j - 100)];
  o[i] = (bf16)(v > 0.f ? v : 0.f);
}
// w[t,b] = sum_a softmax_b(att[t,b,a])   (softmax over the 100 axis, per a)
__global__ __launch_bounds__(256) void k_softmax_colsum(const float* __restrict__ att,
                                                        float* __restrict__ w) {
  const int t = blockIdx.x;
  __shared__ float ws[100];
  for (int b = threadIdx.x; b < 100; b += 256) ws[b] = 0.f;
  __syncthreads();
  const float* base = att + (size_t)t * 100 * 256 + threadIdx.x;
  float mx = -1e30f;
  for (int b = 0; b < 100; ++b) mx = fmaxf(mx, base[b * 256]);
  float s = 0.f;
  for (int b = 0; b < 100; ++b) s += __expf(base[b * 256] - mx);
  const float inv = 1.f / s;
  for (int b = 0; b < 100; ++b) {
    float v = __expf(base[b * 256] - mx) * inv;
    for (int off = 16; off; off >>= 1) v += __shfl_down(v, off, 32);
    if ((threadIdx.x & 31) == 0) atomicAdd(&ws[b], v);
  }
  __syncthreads();
  for (int b = threadIdx.x; b < 100; b += 256) w[(size_t)t * 100 + b] = ws[b];
}
// se[t,d] = (1/R) * sum_b w[t,b] * hcat[t,b,d]   -> (1024, 224) bf16, padded
__global__ __launch_bounds__(256) void k_se_build(const float* __restrict__ w,
                                                  const bf16* __restrict__ hcat,
                                                  bf16* __restrict__ se) {
  const int t = blockIdx.x;
  __shared__ float ws[100];
  for (int b = threadIdx.x; b < 100; b += 256) ws[b] = w[(size_t)t * 100 + b];
  __syncthreads();
  const int d = threadIdx.x;
  if (d < 224) {
    float acc = 0.f;
    if (d < 200) {
      const bf16* base = hcat + (size_t)t * 100 * 224 + d;
      for (int b = 0; b < 100; ++b) acc += ws[b] * (float)base[b * 224];
      acc *= 0.1f;                    // / R, R = 10
    }
    se[(size_t)t * 224 + d] = (bf16)acc;
  }
}
// scatter-add edge messages (84-dim, pre-linear)
__global__ void k_edge_scatter(const float* __restrict__ feat, const int* __restrict__ src,
                               const int* __restrict__ dst, const float* __restrict__ emask,
                               float* __restrict__ agg, long long EF) {
  long long id = blockIdx.x * (long long)blockDim.x + threadIdx.x;
  if (id >= EF) return;
  const int e = (int)(id / FXD), f = (int)(id % FXD);
  float v = feat[(size_t)src[e] * FXD + f];
  if (emask) v *= emask[e];
  atomicAdd(&agg[(size_t)dst[e] * FXD + f], v);
}
// gcat = [max-pool | sum-pool] -> (1024, 1696) bf16, cols 1680..1695 zero
__global__ void k_gcat(const float* __restrict__ mx, const float* __restrict__ sm,
                       bf16* __restrict__ o) {
  int i = blockIdx.x * blockDim.x + threadIdx.x;
  if (i >= NGRAPHS * 1696) return;
  int g = i / 1696, j = i % 1696;
  float v = 0.f;
  if (j < FOUT) v = mx[(size_t)g * FOUT + j];
  else if (j < 2 * FOUT) v = sm[(size_t)g * FOUT + (j - FOUT)];
  o[i] = (bf16)v;
}
__global__ void k_zcat(const float* __restrict__ xg, const float* __restrict__ se,
                       bf16* __restrict__ o) {
  int i = blockIdx.x * blockDim.x + threadIdx.x;
  if (i >= NGRAPHS * 768) return;
  int g = i / 768, j = i % 768;
  float v = j < 512 ? xg[(size_t)g * 512 + j] : se[(size_t)g * 256 + (j - 512)];
  o[i] = (bf16)v;
}
// z[g] = dot(z1[g,:256], g4W) + g4b
__global__ __launch_bounds__(256) void k_final_dot(const bf16* __restrict__ z1,
                                                   const float* __restrict__ g4W,
                                                   const float* __restrict__ g4b,
                                                   float* __restrict__ out) {
  const int g = blockIdx.x;
  float v = (float)z1[(size_t)g * 256 + threadIdx.x] * g4W[threadIdx.x];
  for (int off = 16; off; off >>= 1) v += __shfl_down(v, off, 32);
  __shared__ float ps[8];
  if ((threadIdx.x & 31) == 0) ps[threadIdx.x >> 5] = v;
  __syncthreads();
  if (threadIdx.x == 0) {
    float s = 0.f;
    for (int i = 0; i < 8; ++i) s += ps[i];
    out[g] = s + g4b[0];
  }
}

// ---------------------------------------------------------------------------
// Host-side launchers
// ---------------------------------------------------------------------------
template<int ACT, int OM>
static inline void launch_gemm(const bf16* A, int lda, const bf16* W, int ldw,
                               const float* bias, void* C, int ldc,
                               int M, int N, int Kp, const int* gidx, float* pm,
                               float* ps, int pld, hipStream_t st) {
  dim3 grid((unsigned)((N + 511) / 512), (unsigned)(M / 16));
  gemm_bias_act<ACT, OM><<<grid, 256, 0, st>>>(A, lda, W, ldw, bias, C, ldc,
                                               M, N, Kp, gidx, pm, ps, pld);
}
static inline dim3 g1d(long long n) { return dim3((unsigned)((n + 255) / 256)); }

extern "C" void kernel_launch(void* const* d_in, const int* in_sizes, int n_in,
                              void* d_out, int out_size, void* d_ws, size_t ws_size,
                              hipStream_t stream) {
  (void)in_sizes; (void)n_in; (void)out_size; (void)ws_size;
  // ---- inputs (setup_inputs dict order) ----
  const float* x     = (const float*)d_in[0];
  const int*   ei    = (const int*)d_in[1];
  const int*   src   = ei;
  const int*   dst   = ei + NEDGES;
  const int*   batch = (const int*)d_in[2];
  const float* smi   = (const float*)d_in[3];
  const float* y_x   = (const float*)d_in[4];
  const float* emask = (const float*)d_in[5];
  const float* Wih_f = (const float*)d_in[6];
  const float* Whh_f = (const float*)d_in[7];
  const float* bih_f = (const float*)d_in[8];
  const float* bhh_f = (const float*)d_in[9];
  const float* Wih_b = (const float*)d_in[10];
  const float* Whh_b = (const float*)d_in[11];
  const float* bih_b = (const float*)d_in[12];
  const float* bhh_b = (const float*)d_in[13];
  const float* f1_W  = (const float*)d_in[14];  const float* f1_b  = (const float*)d_in[15];
  const float* f2_W  = (const float*)d_in[16];  const float* f2_b  = (const float*)d_in[17];
  const float* f11_W = (const float*)d_in[18];  const float* f11_b = (const float*)d_in[19];
  const float* f21_W = (const float*)d_in[20];  const float* f21_b = (const float*)d_in[21];
  const float* lin1_W= (const float*)d_in[22];  const float* lin1_b= (const float*)d_in[23];
  const float* lin2_W= (const float*)d_in[24];  const float* lin2_b= (const float*)d_in[25];
  const float* g1_W  = (const float*)d_in[26];  const float* g1_b  = (const float*)d_in[27];
  const float* g2_W  = (const float*)d_in[28];  const float* g2_b  = (const float*)d_in[29];
  const float* g3_W  = (const float*)d_in[30];  const float* g3_b  = (const float*)d_in[31];
  const float* g4_W  = (const float*)d_in[32];  const float* g4_b  = (const float*)d_in[33];

  float* out_z  = (float*)d_out;            // (1024,1)
  float* out_xg = out_z + NGRAPHS;          // (1024,512)
  float* out_yg = out_xg + NGRAPHS * 512;   // (1024,512)

  // ---- workspace bump allocator (256B aligned) ----
  char* ws = (char*)d_ws; size_t off = 0;
  auto alloc = [&](size_t bytes) -> char* {
    char* p = ws + off; off = (off + bytes + 255) & ~(size_t)255; return p;
  };
  bf16*  Xbf   = (bf16*)alloc((size_t)SEQ * BGRU * 128 * 2);  // 26 MB, K-pad 128
  float* gi_f  = (float*)alloc((size_t)SEQ * BGRU * 300 * 4); // 123 MB (later: att)
  float* gi_b  = (float*)alloc((size_t)SEQ * BGRU * 300 * 4); // 123 MB (later: a1)
  bf16*  ysf   = (bf16*)alloc((size_t)SEQ * BGRU * 100 * 2);
  bf16*  ysb   = (bf16*)alloc((size_t)SEQ * BGRU * 100 * 2);
  bf16*  hcat  = (bf16*)alloc((size_t)SEQ * BGRU * 224 * 2);  // 46 MB, K-pad 224
  float* wbuf  = (float*)alloc((size_t)SEQ * 100 * 4);
  bf16*  sebf  = (bf16*)alloc((size_t)SEQ * 224 * 2);
  bf16*  se2bf = (bf16*)alloc((size_t)SEQ * 512 * 2);
  float* seout = (float*)alloc((size_t)SEQ * 256 * 4);
  float* agg   = (float*)alloc((size_t)NNODES * FXD * 4);     // 33.6 MB
  bf16*  tbf   = (bf16*)alloc((size_t)NNODES * FXDP * 2);     // K-pad 96
  float* h1    = (float*)alloc((size_t)NNODES * FXD * 4);
  float* pmx   = (float*)alloc((size_t)NGRAPHS * FOUT * 4);
  float* psm   = (float*)alloc((size_t)NGRAPHS * FOUT * 4);
  bf16*  gcat  = (bf16*)alloc((size_t)NGRAPHS * 1696 * 2);    // K-pad 1696
  bf16*  gh1   = (bf16*)alloc((size_t)NGRAPHS * 1024 * 2);
  bf16*  zin   = (bf16*)alloc((size_t)NGRAPHS * 768 * 2);
  bf16*  z1    = (bf16*)alloc((size_t)NGRAPHS * 256 * 2);
  // bf16 weight copies, K-padded with zeros
  bf16* WihP_f = (bf16*)alloc(300 * 128 * 2);
  bf16* WhhP_f = (bf16*)alloc(300 * 128 * 2);
  bf16* WihP_b = (bf16*)alloc(300 * 128 * 2);
  bf16* WhhP_b = (bf16*)alloc(300 * 128 * 2);
  bf16* f1P    = (bf16*)alloc((size_t)512 * 224 * 2);
  bf16* f2P    = (bf16*)alloc((size_t)256 * 512 * 2);
  bf16* f11P   = (bf16*)alloc((size_t)512 * 224 * 2);
  bf16* f21P   = (bf16*)alloc((size_t)256 * 512 * 2);
  bf16* lin1P  = (bf16*)alloc((size_t)FXD * FXDP * 2);
  bf16* lin2P  = (bf16*)alloc((size_t)FOUT * FXDP * 2);
  bf16* g1P    = (bf16*)alloc((size_t)1024 * 1696 * 2);
  bf16* g2P    = (bf16*)alloc((size_t)512 * 1024 * 2);
  bf16* g3P    = (bf16*)alloc((size_t)256 * 768 * 2);
  // dead-buffer aliases (gi_* are only read during the GRU, before these)
  bf16*  a1  = (bf16*)gi_b;   // (102400,512) bf16 = 105 MB <= 123 MB
  float* att = gi_f;          // (102400,256) f32  = 105 MB <= 123 MB

  auto cvt = [&](const float* s, bf16* d, long long rows, int K, int Kp) {
    k_cvt_pad<<<g1d(rows * Kp), 256, 0, stream>>>(s, d, rows, K, Kp);
  };

  // ---- weight + activation conversions (zero-padded K) ----
  cvt(smi, Xbf, (long long)SEQ * BGRU, 100, 128);
  cvt(Wih_f, WihP_f, 300, 100, 128);  cvt(Whh_f, WhhP_f, 300, 100, 128);
  cvt(Wih_b, WihP_b, 300, 100, 128);  cvt(Whh_b, WhhP_b, 300, 100, 128);
  cvt(f1_W, f1P, 512, 200, 224);   cvt(f2_W, f2P, 256, 512, 512);
  cvt(f11_W, f11P, 512, 200, 224); cvt(f21_W, f21P, 256, 512, 512);
  cvt(lin1_W, lin1P, FXD, FXD, FXDP); cvt(lin2_W, lin2P, FOUT, FXD, FXDP);
  cvt(g1_W, g1P, 1024, 1680, 1696);   cvt(g2_W, g2P, 512, 1024, 1024);
  cvt(g3_W, g3P, 256, 768, 768);

  // ---- SMILES branch: input-side GEMMs then persistent fused GRU ----
  launch_gemm<0,0>(Xbf, 128, WihP_f, 128, bih_f, gi_f, 300, SEQ * BGRU, 300, 128,
                   nullptr, nullptr, nullptr, 0, stream);
  launch_gemm<0,0>(Xbf, 128, WihP_b, 128, bih_b, gi_b, 300, SEQ * BGRU, 300, 128,
                   nullptr, nullptr, nullptr, 0, stream);
  (void)hipFuncSetAttribute((const void*)gru_fused,
                            hipFuncAttributeMaxDynamicSharedMemorySize, GRU_LDS_BYTES);
  gru_fused<<<2, 1024, GRU_LDS_BYTES, stream>>>(WhhP_f, WhhP_b, bhh_f, bhh_b,
                                                gi_f, gi_b, ysf, ysb);
  k_hcat<<<g1d((long long)SEQ * BGRU * 224), 256, 0, stream>>>(ysf, ysb, hcat,
                                                               (long long)SEQ * BGRU);
  // attention: a1 = tanh(hcat@f1^T+b), att = a1@f2^T+b
  launch_gemm<2,1>(hcat, 224, f1P, 224, f1_b, a1, 512, SEQ * BGRU, 512, 224,
                   nullptr, nullptr, nullptr, 0, stream);
  launch_gemm<0,0>(a1, 512, f2P, 512, f2_b, att, 256, SEQ * BGRU, 256, 512,
                   nullptr, nullptr, nullptr, 0, stream);
  // emb einsum + sum over 'a' collapses to column-sum of softmax + weighted sum
  k_softmax_colsum<<<SEQ, 256, 0, stream>>>(att, wbuf);
  k_se_build<<<SEQ, 256, 0, stream>>>(wbuf, hcat, sebf);
  launch_gemm<0,1>(sebf, 224, f11P, 224, f11_b, se2bf, 512, SEQ, 512, 224,
                   nullptr, nullptr, nullptr, 0, stream);
  launch_gemm<0,0>(se2bf, 512, f21P, 512, f21_b, seout, 256, SEQ, 256, 512,
                   nullptr, nullptr, nullptr, 0, stream);

  // ---- graph branches (original, then augmented) ----
  const long long EF = (long long)NEDGES * FXD;
  const long long NF = (long long)NNODES * FXD;
  for (int br = 0; br < 2; ++br) {
    const float* xin = br ? y_x : x;
    const float* em  = br ? emask : nullptr;
    float* outg = br ? out_yg : out_xg;
    // GIN1: agg = segment_sum(xin[src]); h1 = relu((xin+agg)@lin1^T+b)
    k_fill_f32<<<g1d(NF), 256, 0, stream>>>(agg, 0.f, NF);
    k_edge_scatter<<<g1d(EF), 256, 0, stream>>>(xin, src, dst, em, agg, EF);
    k_add_pad<<<g1d((long long)NNODES * FXDP), 256, 0, stream>>>(xin, agg, tbf, NNODES);
    launch_gemm<1,0>(tbf, FXDP, lin1P, FXDP, lin1_b, h1, FXD, NNODES, FXD, FXDP,
                     nullptr, nullptr, nullptr, 0, stream);
    // GIN2 + fused max/sum pooling in the GEMM epilogue (h2 never materialized)
    k_fill_f32<<<g1d(NF), 256, 0, stream>>>(agg, 0.f, NF);
    k_edge_scatter<<<g1d(EF), 256, 0, stream>>>(h1, src, dst, em, agg, EF);
    k_add_pad<<<g1d((long long)NNODES * FXDP), 256, 0, stream>>>(h1, agg, tbf, NNODES);
    k_fill_f32<<<g1d((long long)NGRAPHS * FOUT), 256, 0, stream>>>(pmx, 0.f,
                                                        (long long)NGRAPHS * FOUT);
    k_fill_f32<<<g1d((long long)NGRAPHS * FOUT), 256, 0, stream>>>(psm, 0.f,
                                                        (long long)NGRAPHS * FOUT);
    launch_gemm<1,2>(tbf, FXDP, lin2P, FXDP, lin2_b, nullptr, 0, NNODES, FOUT, FXDP,
                     batch, pmx, psm, FOUT, stream);
    // head: g = [mx|sm]; relu(g@g1^T+b1) @ g2^T + b2 -> out (1024,512)
    k_gcat<<<g1d((long long)NGRAPHS * 1696), 256, 0, stream>>>(pmx, psm, gcat);
    launch_gemm<1,1>(gcat, 1696, g1P, 1696, g1_b, gh1, 1024, NGRAPHS, 1024, 1696,
                     nullptr, nullptr, nullptr, 0, stream);
    launch_gemm<0,0>(gh1, 1024, g2P, 1024, g2_b, outg, 512, NGRAPHS, 512, 1024,
                     nullptr, nullptr, nullptr, 0, stream);
  }

  // ---- fusion head ----
  k_zcat<<<g1d((long long)NGRAPHS * 768), 256, 0, stream>>>(out_xg, seout, zin);
  launch_gemm<1,1>(zin, 768, g3P, 768, g3_b, z1, 256, NGRAPHS, 256, 768,
                   nullptr, nullptr, nullptr, 0, stream);
  k_final_dot<<<NGRAPHS, 256, 0, stream>>>(z1, g4_W, g4_b, out_z);
}
